// TransformerBlock_16200616641157
// MI455X (gfx1250) — compile-verified, matrix operand order checked by reading, not run on hardware
//
#include <hip/hip_runtime.h>
#include <hip/hip_bf16.h>
#include <math.h>

typedef _Float16 f16_t;
typedef __attribute__((ext_vector_type(16))) _Float16 v16h;
typedef __attribute__((ext_vector_type(8)))  _Float16 v8h;
typedef __attribute__((ext_vector_type(8)))  float    v8f;

constexpr int cB   = 2;
constexpr int cS   = 2048;
constexpr int cD   = 768;
constexpr int cH   = 12;
constexpr int cDH  = 64;
constexpr int cF   = 3072;
constexpr int cTok = cB * cS;        // 4096
constexpr int cQKV = 3 * cD;         // 2304
constexpr int c2D  = 2 * cD;         // 1536

// ---------------------------------------------------------------------------
// CDNA5 async global->LDS copies (ASYNCcnt-tracked, per CDNA5 ISA)
// LDS byte address = low 32 bits of flat shared pointer (aperture rule).
// Async loads complete in order -> partial waits enable double buffering.
// ---------------------------------------------------------------------------
__device__ __forceinline__ unsigned lds_off_of(const void* p) {
  return (unsigned)(unsigned long long)p;
}
__device__ __forceinline__ void async_copy_b128(const void* lds_dst,
                                                const void* gsrc) {
  asm volatile("global_load_async_to_lds_b128 %0, %1, off"
               :: "v"(lds_off_of(lds_dst)), "v"((unsigned long long)gsrc)
               : "memory");
}
__device__ __forceinline__ void async_wait0() {
  asm volatile("s_wait_asynccnt 0x0" ::: "memory");
}
__device__ __forceinline__ void async_wait3() {
  asm volatile("s_wait_asynccnt 0x3" ::: "memory");
}
__device__ __forceinline__ void async_wait4() {
  asm volatile("s_wait_asynccnt 0x4" ::: "memory");
}

// ---------------------------------------------------------------------------
// Fragment helpers (CDNA5 16-bit WMMA layouts, wave32)
// A (16x32): lane = h*16 + m, pair (2v,2v+1) -> K = ((v&3)<<1)+(h<<3)+((v>>2)<<4)
// B (32x16): mirrors A with N in place of M
// C (16x16 f32): lane = h*16 + n, VGPR r -> M = r + 8*h
// ---------------------------------------------------------------------------
__device__ __forceinline__ int pair_k(int v, int h) {
  return ((v & 3) << 1) + (h << 3) + ((v >> 2) << 4);
}

// rows of `base` are the 16-wide dim, pair dim contiguous: base[m*ld + k]
__device__ __forceinline__ v16h load_frag_kcontig(const f16_t* base, int ld) {
  const int lane = threadIdx.x & 31;
  const int m = lane & 15, hh = lane >> 4;
  v16h r;
#pragma unroll
  for (int v = 0; v < 8; ++v) {
    const int k0 = pair_k(v, hh);
    r[2 * v]     = base[m * ld + k0];
    r[2 * v + 1] = base[m * ld + k0 + 1];
  }
  return r;
}

// element (k, n) at base[k*ld + n] (pair dim strided): for B fragments
__device__ __forceinline__ v16h load_frag_kstrided(const f16_t* base, int ld) {
  const int lane = threadIdx.x & 31;
  const int n = lane & 15, hh = lane >> 4;
  v16h r;
#pragma unroll
  for (int v = 0; v < 8; ++v) {
    const int k0 = pair_k(v, hh);
    r[2 * v]     = base[k0 * ld + n];
    r[2 * v + 1] = base[(k0 + 1) * ld + n];
  }
  return r;
}

// ---------------------------------------------------------------------------
// f32 -> f16 conversion (weights)
// ---------------------------------------------------------------------------
__global__ void cvt_f16_kernel(const float* __restrict__ in,
                               f16_t* __restrict__ out, int n) {
  for (int i = blockIdx.x * blockDim.x + threadIdx.x; i < n;
       i += gridDim.x * blockDim.x)
    out[i] = (f16_t)in[i];
}

// ---------------------------------------------------------------------------
// ss[b][j] = bias[j] + sum_d silu(t[b,d]) * w[d][j]
// ---------------------------------------------------------------------------
__global__ __launch_bounds__(256) void cond_kernel(
    const float* __restrict__ tv, const float* __restrict__ w,
    const float* __restrict__ bias, float* __restrict__ ss) {
  const int j = blockIdx.x * 256 + threadIdx.x;
  const int b = blockIdx.y;
  float acc = bias[j];
  for (int d = 0; d < cD; ++d) {
    const float t = tv[b * cD + d];
    const float sil = t / (1.0f + __expf(-t));
    acc += sil * w[(size_t)d * c2D + j];
  }
  ss[b * c2D + j] = acc;
}

// ---------------------------------------------------------------------------
// AdaLN: LayerNorm (no affine) then (1+scale)*h + bias, output f16
// ---------------------------------------------------------------------------
__global__ __launch_bounds__(256) void adaln_kernel(
    const float* __restrict__ x, const float* __restrict__ ss,
    f16_t* __restrict__ out) {
  __shared__ float red[2][8];
  const int t = blockIdx.x;
  const int b = t / cS;
  const int tid = threadIdx.x;
  const float* xr = x + (size_t)t * cD;
  const float v0 = xr[tid], v1 = xr[tid + 256], v2 = xr[tid + 512];
  float s = v0 + v1 + v2;
  float q = v0 * v0 + v1 * v1 + v2 * v2;
#pragma unroll
  for (int off = 16; off > 0; off >>= 1) {
    s += __shfl_xor(s, off, 32);
    q += __shfl_xor(q, off, 32);
  }
  const int wave = tid >> 5, lane = tid & 31;
  if (lane == 0) { red[0][wave] = s; red[1][wave] = q; }
  __syncthreads();
  float st = 0.f, qt = 0.f;
#pragma unroll
  for (int w = 0; w < 8; ++w) { st += red[0][w]; qt += red[1][w]; }
  const float mean = st * (1.0f / cD);
  const float var  = qt * (1.0f / cD) - mean * mean;
  const float rstd = rsqrtf(var + 1e-5f);
  const float* ssb = ss + (size_t)b * c2D;
  f16_t* orow = out + (size_t)t * cD;
#pragma unroll
  for (int j = 0; j < 3; ++j) {
    const int d = tid + j * 256;
    const float nrm = (xr[d] - mean) * rstd;
    orow[d] = (f16_t)((1.0f + ssb[d]) * nrm + ssb[cD + d]);
  }
}

// ---------------------------------------------------------------------------
// Tiled WMMA GEMM: block tile 128x64, BK=32, 8 waves, wave tile 32x32.
// Double-buffered LDS; tile t+1 staged via async b128 while computing tile t.
// ---------------------------------------------------------------------------
enum { EPI_F16_BIAS = 0, EPI_F16_GELU = 1, EPI_F32_RES = 2 };

template <int EPI>
__global__ __launch_bounds__(256) void gemm_wmma_kernel(
    const f16_t* __restrict__ A, const f16_t* __restrict__ Bm,
    const float* __restrict__ bias, const float* __restrict__ res,
    f16_t* __restrict__ outH, float* __restrict__ outF,
    int M, int N, int K) {
  __shared__ __align__(16) f16_t As[2][128][40];  // [m][k] padded (80B rows)
  __shared__ __align__(16) f16_t Bs[2][32][72];   // [k][n] padded (144B rows)

  const int n0 = blockIdx.x * 64;
  const int m0 = blockIdx.y * 128;
  const int tid = threadIdx.x;
  const int wave = tid >> 5;
  const int lane = tid & 31;
  const int wr = wave & 3;   // 4 row groups of 32
  const int wc = wave >> 2;  // 2 col groups of 32

  v8f c00 = {}, c01 = {}, c10 = {}, c11 = {};

  // staging coordinates (3 async b128 per thread per tile)
  const int arow = tid >> 1;            // 0..127
  const int acol = (tid & 1) << 4;      // 0 or 16
  const int bk   = tid >> 3;            // 0..31
  const int bn   = (tid & 7) << 3;      // 0..56

  auto stage = [&](int buf, int k0) {
    const f16_t* ga = A + (size_t)(m0 + arow) * K + k0 + acol;
    async_copy_b128(&As[buf][arow][acol], ga);
    async_copy_b128(&As[buf][arow][acol + 8], ga + 8);
    async_copy_b128(&Bs[buf][bk][bn], Bm + (size_t)(k0 + bk) * N + n0 + bn);
  };

  const int ksteps = K >> 5;
  stage(0, 0);
  for (int ks = 0; ks < ksteps; ++ks) {
    const int buf = ks & 1;
    const bool more = (ks + 1) < ksteps;
    if (more) stage(buf ^ 1, (ks + 1) << 5);   // prefetch next tile
    if (more) async_wait3(); else async_wait0();  // current tile landed
    __syncthreads();

    const v16h a0 = load_frag_kcontig(&As[buf][wr * 32][0], 40);
    const v16h a1 = load_frag_kcontig(&As[buf][wr * 32 + 16][0], 40);
    const v16h b0 = load_frag_kstrided(&Bs[buf][0][wc * 32], 72);
    const v16h b1 = load_frag_kstrided(&Bs[buf][0][wc * 32 + 16], 72);
    c00 = __builtin_amdgcn_wmma_f32_16x16x32_f16(false, a0, false, b0,
                                                 (short)0, c00, false, false);
    c01 = __builtin_amdgcn_wmma_f32_16x16x32_f16(false, a0, false, b1,
                                                 (short)0, c01, false, false);
    c10 = __builtin_amdgcn_wmma_f32_16x16x32_f16(false, a1, false, b0,
                                                 (short)0, c10, false, false);
    c11 = __builtin_amdgcn_wmma_f32_16x16x32_f16(false, a1, false, b1,
                                                 (short)0, c11, false, false);
    __syncthreads();  // reads done before this buffer is re-staged
  }

  const int n_in = lane & 15, hc = lane >> 4;
#pragma unroll
  for (int ar = 0; ar < 2; ++ar) {
#pragma unroll
    for (int ac = 0; ac < 2; ++ac) {
      const v8f& c = (ar == 0) ? (ac == 0 ? c00 : c01) : (ac == 0 ? c10 : c11);
      const int col = n0 + wc * 32 + ac * 16 + n_in;
#pragma unroll
      for (int r = 0; r < 8; ++r) {
        const int row = m0 + wr * 32 + ar * 16 + r + 8 * hc;
        float v = c[r] + bias[col];
        const size_t idx = (size_t)row * N + col;
        if (EPI == EPI_F16_BIAS) {
          outH[idx] = (f16_t)v;
        } else if (EPI == EPI_F16_GELU) {
          v = 0.5f * v * (1.0f + erff(v * 0.70710678118654752f));
          outH[idx] = (f16_t)v;
        } else {
          outF[idx] = v + res[idx];
        }
      }
    }
  }
}

// ---------------------------------------------------------------------------
// Flash attention: grid (S/64, H, B), 128 threads (4 waves x 16 queries).
// K/V tiles (32 keys x 64) double-buffered in LDS via async b128.
// ---------------------------------------------------------------------------
__global__ __launch_bounds__(128) void attn_kernel(
    const f16_t* __restrict__ qkv, const float* __restrict__ attn_mask,
    const int* __restrict__ kpm, f16_t* __restrict__ out) {
  __shared__ __align__(16) f16_t Ks[2][32][72];   // [key][d]
  __shared__ __align__(16) f16_t Vs[2][32][72];   // [key][d]
  __shared__ f16_t Pld[4][16][32];

  const int tid = threadIdx.x;
  const int wave = tid >> 5;
  const int lane = tid & 31;
  const int m = lane & 15, hh = lane >> 4;  // A-frag indexing
  const int n = m, hc = hh;                 // B-frag / C indexing

  const int h = blockIdx.y;
  const int b = blockIdx.z;
  const int q0 = blockIdx.x * 64 + wave * 16;

  // staging coords: thread -> (key row, 16-wide d group); 4 async b128/thread
  const int skd = tid & 31;
  const int sg  = (tid >> 5) << 4;  // 0,16,32,48

  auto stage = [&](int buf, int kb) {
    const f16_t* kg =
        qkv + (size_t)(b * cS + kb + skd) * cQKV + cD + h * cDH + sg;
    const f16_t* vg = kg + cD;  // V is +768 halves after K within a row
    async_copy_b128(&Ks[buf][skd][sg], kg);
    async_copy_b128(&Ks[buf][skd][sg + 8], kg + 8);
    async_copy_b128(&Vs[buf][skd][sg], vg);
    async_copy_b128(&Vs[buf][skd][sg + 8], vg + 8);
  };

  // Q fragments, 1/sqrt(DH)=0.125 folded in
  const f16_t* qrow = qkv + (size_t)(b * cS + q0 + m) * cQKV + h * cDH;
  v16h qa0, qa1;
#pragma unroll
  for (int v = 0; v < 8; ++v) {
    const int k0 = pair_k(v, hh);
    qa0[2 * v]     = (f16_t)((float)qrow[k0] * 0.125f);
    qa0[2 * v + 1] = (f16_t)((float)qrow[k0 + 1] * 0.125f);
    qa1[2 * v]     = (f16_t)((float)qrow[32 + k0] * 0.125f);
    qa1[2 * v + 1] = (f16_t)((float)qrow[32 + k0 + 1] * 0.125f);
  }

  float mi[8], li[8];
  v8f o[4] = {};
#pragma unroll
  for (int r = 0; r < 8; ++r) { mi[r] = -1e30f; li[r] = 0.f; }

  constexpr int ktiles = cS / 32;
  stage(0, 0);
  for (int kt = 0; kt < ktiles; ++kt) {
    const int buf = kt & 1;
    const int kb = kt * 32;
    const bool more = (kt + 1) < ktiles;
    if (more) stage(buf ^ 1, kb + 32);            // prefetch next K/V tile
    if (more) async_wait4(); else async_wait0();  // current tile landed
    __syncthreads();

    // ---- scores: two 16x16 tiles over 32 keys (K pairs contiguous) ----
    v8f s0 = {}, s1 = {};
    {
      const v16h kf00 = load_frag_kcontig(&Ks[buf][0][0], 72);
      const v16h kf01 = load_frag_kcontig(&Ks[buf][0][32], 72);
      const v16h kf10 = load_frag_kcontig(&Ks[buf][16][0], 72);
      const v16h kf11 = load_frag_kcontig(&Ks[buf][16][32], 72);
      s0 = __builtin_amdgcn_wmma_f32_16x16x32_f16(false, qa0, false, kf00,
                                                  (short)0, s0, false, false);
      s0 = __builtin_amdgcn_wmma_f32_16x16x32_f16(false, qa1, false, kf01,
                                                  (short)0, s0, false, false);
      s1 = __builtin_amdgcn_wmma_f32_16x16x32_f16(false, qa0, false, kf10,
                                                  (short)0, s1, false, false);
      s1 = __builtin_amdgcn_wmma_f32_16x16x32_f16(false, qa1, false, kf11,
                                                  (short)0, s1, false, false);
    }

    // ---- masks + online softmax ----
    const bool msk0 = kpm[b * cS + kb + n] != 0;
    const bool msk1 = kpm[b * cS + kb + 16 + n] != 0;
#pragma unroll
    for (int r = 0; r < 8; ++r) {
      const int qq = q0 + r + 8 * hc;
      float e0 = s0[r] + attn_mask[(size_t)qq * cS + kb + n];
      float e1 = s1[r] + attn_mask[(size_t)qq * cS + kb + 16 + n];
      if (msk0) e0 = -1e30f;
      if (msk1) e1 = -1e30f;
      float mx = fmaxf(e0, e1);
#pragma unroll
      for (int off = 1; off < 16; off <<= 1)
        mx = fmaxf(mx, __shfl_xor(mx, off, 32));
      const float mnew = fmaxf(mi[r], mx);
      const float corr = __expf(mi[r] - mnew);
      const float p0 = __expf(e0 - mnew);
      const float p1 = __expf(e1 - mnew);
      float ps = p0 + p1;
#pragma unroll
      for (int off = 1; off < 16; off <<= 1) ps += __shfl_xor(ps, off, 32);
      li[r] = li[r] * corr + ps;
      mi[r] = mnew;
#pragma unroll
      for (int g = 0; g < 4; ++g) o[g][r] *= corr;
      Pld[wave][r + 8 * hc][n]      = (f16_t)p0;
      Pld[wave][r + 8 * hc][16 + n] = (f16_t)p1;
    }
    // Pld region is per-wave; LDS is in-order within a wave — only block
    // compiler reordering here.
    __builtin_amdgcn_wave_barrier();

    // P: C-layout -> A-fragment via LDS
    v16h pa;
#pragma unroll
    for (int v = 0; v < 8; ++v) {
      const int k0 = pair_k(v, hh);
      pa[2 * v]     = Pld[wave][m][k0];
      pa[2 * v + 1] = Pld[wave][m][k0 + 1];
    }

    // ---- O += P @ V (4 column groups of 16 over DH=64) ----
#pragma unroll
    for (int g = 0; g < 4; ++g) {
      const v16h vf = load_frag_kstrided(&Vs[buf][0][g * 16], 72);
      o[g] = __builtin_amdgcn_wmma_f32_16x16x32_f16(false, pa, false, vf,
                                                    (short)0, o[g], false,
                                                    false);
    }
    __syncthreads();  // all reads done before this buffer is re-staged
  }

  // ---- normalize and write [token][h*64 + d] ----
#pragma unroll
  for (int r = 0; r < 8; ++r) {
    const float inv = 1.0f / li[r];
    const size_t base =
        (size_t)(b * cS + q0 + r + 8 * hc) * cD + h * cDH + n;
#pragma unroll
    for (int g = 0; g < 4; ++g)
      out[base + g * 16] = (f16_t)(o[g][r] * inv);
  }
}

// ---------------------------------------------------------------------------
// Host: orchestrate the block
// ---------------------------------------------------------------------------
extern "C" void kernel_launch(void* const* d_in, const int* in_sizes, int n_in,
                              void* d_out, int out_size, void* d_ws,
                              size_t ws_size, hipStream_t stream) {
  (void)in_sizes; (void)n_in; (void)out_size; (void)ws_size;
  const float* x     = (const float*)d_in[0];
  const float* amask = (const float*)d_in[1];
  const int*   kpm   = (const int*)d_in[2];
  const float* tvec  = (const float*)d_in[3];
  const float* Wqkv  = (const float*)d_in[4];
  const float* bqkv  = (const float*)d_in[5];
  const float* Wproj = (const float*)d_in[6];
  const float* bproj = (const float*)d_in[7];
  const float* n1w   = (const float*)d_in[8];
  const float* n1b   = (const float*)d_in[9];
  const float* n2w   = (const float*)d_in[10];
  const float* n2b   = (const float*)d_in[11];
  const float* Wm1   = (const float*)d_in[12];
  const float* bm1   = (const float*)d_in[13];
  const float* Wm2   = (const float*)d_in[14];
  const float* bm2   = (const float*)d_in[15];
  float* out = (float*)d_out;

  char* ws = (char*)d_ws;
  size_t off = 0;
  auto alloc = [&](size_t bytes) -> void* {
    void* p = ws + off;
    off += (bytes + 255) & ~(size_t)255;
    return p;
  };
  f16_t* Wqkv16  = (f16_t*)alloc((size_t)cD * cQKV * 2);
  f16_t* Wproj16 = (f16_t*)alloc((size_t)cD * cD * 2);
  f16_t* Wm1_16  = (f16_t*)alloc((size_t)cD * cF * 2);
  f16_t* Wm2_16  = (f16_t*)alloc((size_t)cF * cD * 2);
  f16_t* h16     = (f16_t*)alloc((size_t)cTok * cD * 2);
  f16_t* qkv16   = (f16_t*)alloc((size_t)cTok * cQKV * 2);
  f16_t* attn16  = (f16_t*)alloc((size_t)cTok * cD * 2);
  f16_t* act16   = (f16_t*)alloc((size_t)cTok * cF * 2);
  float* x1      = (float*)alloc((size_t)cTok * cD * 4);
  float* ss1     = (float*)alloc((size_t)cB * c2D * 4);
  float* ss2     = (float*)alloc((size_t)cB * c2D * 4);

  // weight conversion f32 -> f16
  cvt_f16_kernel<<<1024, 256, 0, stream>>>(Wqkv, Wqkv16, cD * cQKV);
  cvt_f16_kernel<<<512, 256, 0, stream>>>(Wproj, Wproj16, cD * cD);
  cvt_f16_kernel<<<1024, 256, 0, stream>>>(Wm1, Wm1_16, cD * cF);
  cvt_f16_kernel<<<1024, 256, 0, stream>>>(Wm2, Wm2_16, cF * cD);

  // conditioning: ss = silu(t) @ n_w + n_b
  cond_kernel<<<dim3(c2D / 256, cB), 256, 0, stream>>>(tvec, n1w, n1b, ss1);
  cond_kernel<<<dim3(c2D / 256, cB), 256, 0, stream>>>(tvec, n2w, n2b, ss2);

  // adaLN 1 -> h16
  adaln_kernel<<<cTok, 256, 0, stream>>>(x, ss1, h16);

  // qkv = h @ Wqkv + bqkv  -> f16 [tok][3D]
  gemm_wmma_kernel<EPI_F16_BIAS>
      <<<dim3(cQKV / 64, cTok / 128), 256, 0, stream>>>(
          h16, Wqkv16, bqkv, nullptr, qkv16, nullptr, cTok, cQKV, cD);

  // flash attention -> attn16 [tok][D]
  attn_kernel<<<dim3(cS / 64, cH, cB), 128, 0, stream>>>(qkv16, amask, kpm,
                                                         attn16);

  // x1 = x + attn @ Wproj + bproj  (f32)
  gemm_wmma_kernel<EPI_F32_RES>
      <<<dim3(cD / 64, cTok / 128), 256, 0, stream>>>(
          attn16, Wproj16, bproj, x, nullptr, x1, cTok, cD, cD);

  // adaLN 2 -> h16 (reuse)
  adaln_kernel<<<cTok, 256, 0, stream>>>(x1, ss2, h16);

  // act = gelu(h2 @ Wm1 + bm1) -> f16
  gemm_wmma_kernel<EPI_F16_GELU>
      <<<dim3(cF / 64, cTok / 128), 256, 0, stream>>>(
          h16, Wm1_16, bm1, nullptr, act16, nullptr, cTok, cF, cD);

  // out = x1 + act @ Wm2 + bm2 (f32)
  gemm_wmma_kernel<EPI_F32_RES>
      <<<dim3(cD / 64, cTok / 128), 256, 0, stream>>>(
          act16, Wm2_16, bm2, x1, nullptr, out, cTok, cD, cF);
}
